// Basic_Transformer_86612310491180
// MI455X (gfx1250) — matrix operand order repaired and checked by reading, not compiled, not run on hardware
//
#include <hip/hip_runtime.h>
#include <hip/hip_bf16.h>

// ---------------------------------------------------------------------------
// Basic transformer (B=2,S=2048,C=512,L=4,H=8,CH=64) for gfx1250 (MI455X).
// FP16 WMMA (V_WMMA_F32_16X16X32_F16) with FP32 accumulation.
// GEMMs: LDS-staged block tiles filled by the Tensor Data Mover
// (TENSOR_LOAD_TO_LDS + s_wait_tensorcnt) when available.
// ---------------------------------------------------------------------------

#define BB   2
#define SS   2048
#define CC   512
#define LL   4
#define HH   8
#define CHD  64
#define ODD  256          // 4*CH
#define ROWS (BB * SS)    // 4096

typedef __attribute__((ext_vector_type(16))) _Float16 v16h;
typedef __attribute__((ext_vector_type(8)))  _Float16 v8h;
typedef __attribute__((ext_vector_type(8)))  float    v8f;
typedef unsigned __attribute__((ext_vector_type(4))) u32x4;
typedef int      __attribute__((ext_vector_type(4))) i32x4;
typedef int      __attribute__((ext_vector_type(8))) i32x8;

#if __has_builtin(__builtin_amdgcn_tensor_load_to_lds) && \
    __has_builtin(__builtin_amdgcn_s_wait_tensorcnt)
#define TDM_AVAILABLE 1
#else
#define TDM_AVAILABLE 0
#endif

// ---------------------------------------------------------------------------
// WMMA helpers
// ---------------------------------------------------------------------------
__device__ __forceinline__ v8f wmma_f16(v16h a, v16h b, v8f c) {
  return __builtin_amdgcn_wmma_f32_16x16x32_f16(
      /*neg_a=*/false, a, /*neg_b=*/false, b,
      /*c_mod=*/(short)0, c, /*reuse_a=*/false, /*reuse_b=*/false);
}

// A-fragment (16x32, M x K), row-major source with leading dim `ld` (halves).
// Lane L holds row M=L&15; lanes 0-15: K = {0..7, 16..23},
// lanes 16-31: K = {8..15, 24..31}.
__device__ __forceinline__ v16h load_fragA(const _Float16* __restrict__ base, int ld) {
  const int lane = threadIdx.x & 31;
  const int row  = lane & 15;
  const int k0   = (lane >> 4) << 3;                    // 0 or 8
  const _Float16* p = base + (size_t)row * ld + k0;
  v8h lo = *(const v8h*)(p);
  v8h hi = *(const v8h*)(p + 16);
  v16h r;
#pragma unroll
  for (int i = 0; i < 8; ++i) { r[i] = lo[i]; r[i + 8] = hi[i]; }
  return r;
}

// B-fragment (32x16, K x N) supplied as B^T (N x K row-major, leading dim ld).
// Lane L holds column N=L&15; lanes 0-15: K = {0..7, 16..23}, lanes 16-31: K = {8..15, 24..31}.
__device__ __forceinline__ v16h load_fragB(const _Float16* __restrict__ bt, int ld) {
  const int lane = threadIdx.x & 31;
  const int n    = lane & 15;
  const int k0   = (lane >> 4) << 3;                    // 0 or 8
  const _Float16* p = bt + (size_t)n * ld + k0;
  v8h lo = *(const v8h*)(p);
  v8h hi = *(const v8h*)(p + 16);
  v16h r;
#pragma unroll
  for (int i = 0; i < 8; ++i) { r[i] = lo[i]; r[i + 8] = hi[i]; }
  return r;
}

// ---------------------------------------------------------------------------
// Tensor Data Mover: 2D tile (rows x cols f16 elements, global row stride ld)
// -> contiguous LDS (rows*cols halves, row-major, no padding).
// D# layout per CDNA5 ISA.
// ---------------------------------------------------------------------------
__device__ __forceinline__ void tdm_load_tile_2d(const _Float16* gsrc,
                                                 unsigned lds_off,
                                                 int rows, int cols, int ld) {
#if TDM_AVAILABLE
  unsigned long long ga = (unsigned long long)gsrc;
  // group0: count=1 | lds_addr | global_addr[56:0] | type=2 in bits 127:126
  u32x4 g0 = { 1u,
               lds_off,
               (unsigned)(ga & 0xffffffffu),
               (unsigned)((ga >> 32) & 0x01ffffffu) | 0x80000000u };
  // group1: data_size=2B; tensor_dim0=ld; tensor_dim1=rows;
  //         tile_dim0=cols; tile_dim1=rows; tile_dim2=0;
  //         tensor_dim0_stride=ld; tensor_dim1_stride=0.
  i32x8 g1 = { (int)(1u << 16),
               (int)(((unsigned)ld & 0xffffu) << 16),
               (int)((((unsigned)ld >> 16) & 0xffffu) |
                     (((unsigned)rows & 0xffffu) << 16)),
               (int)(((unsigned)cols & 0xffffu) << 16),
               rows,
               ld,
               0, 0 };
  i32x4 z4 = { 0, 0, 0, 0 };
#if __clang_major__ >= 23
  i32x8 z8 = { 0, 0, 0, 0, 0, 0, 0, 0 };
  __builtin_amdgcn_tensor_load_to_lds(g0, g1, z4, z4, z8, 0);
#else
  __builtin_amdgcn_tensor_load_to_lds(g0, g1, z4, z4, 0);
#endif
#else
  (void)gsrc; (void)lds_off; (void)rows; (void)cols; (void)ld;
#endif
}

// ---------------------------------------------------------------------------
// Weight convert + transpose: W (R x N, f32 row-major) -> Wt (N x R, f16)
// ---------------------------------------------------------------------------
__global__ void cvt_transpose_kernel(const float* __restrict__ W,
                                     _Float16* __restrict__ Wt,
                                     int R, int N) {
  int t = blockIdx.x * blockDim.x + threadIdx.x;
  if (t >= R * N) return;
  int r = t / N, c = t % N;
  Wt[(size_t)c * R + r] = (_Float16)W[(size_t)r * N + c];
}

// ---------------------------------------------------------------------------
// LayerNorm: one wave per row of 512; writes f16 activations.
// ---------------------------------------------------------------------------
__global__ __launch_bounds__(256) void ln_kernel(const float* __restrict__ X,
                                                 const float* __restrict__ gamma,
                                                 const float* __restrict__ beta,
                                                 _Float16* __restrict__ Hf) {
  const int wave = threadIdx.x >> 5;
  const int lane = threadIdx.x & 31;
  const int row  = blockIdx.x * 8 + wave;
  const float* xr = X + (size_t)row * CC;
  float vals[16];
  float s = 0.f;
#pragma unroll
  for (int i = 0; i < 16; ++i) { vals[i] = xr[lane + i * 32]; s += vals[i]; }
#pragma unroll
  for (int d = 1; d < 32; d <<= 1) s += __shfl_xor(s, d, 32);
  float mu = s * (1.0f / (float)CC);
  float v = 0.f;
#pragma unroll
  for (int i = 0; i < 16; ++i) { float t = vals[i] - mu; v += t * t; }
#pragma unroll
  for (int d = 1; d < 32; d <<= 1) v += __shfl_xor(v, d, 32);
  float rstd = rsqrtf(v * (1.0f / (float)CC) + 1e-5f);
  _Float16* hr = Hf + (size_t)row * CC;
#pragma unroll
  for (int i = 0; i < 16; ++i) {
    int c = lane + i * 32;
    hr[c] = (_Float16)((vals[i] - mu) * rstd * gamma[c] + beta[c]);
  }
}

// ---------------------------------------------------------------------------
// LDS-staged WMMA GEMM: C[M,N] = A[M,K] * B[K,N] (+bias +residual).
// Bt is B transposed (N x K row-major). Block = 256 threads (8 waves).
// Block tile 128(M) x 64(N); per 32-wide k-step, A-tile (128x32, 8KB) and
// B-tile (64x32, 4KB) are DMA'd into LDS by the TDM (fallback: manual copy).
// Wave grid 4x2; each wave computes 32x32 = 4 wmma per k-step from LDS.
// ---------------------------------------------------------------------------
__global__ __launch_bounds__(256) void gemm_wmma_kernel(
    const _Float16* __restrict__ A,   // M x K row-major f16
    const _Float16* __restrict__ Bt,  // N x K row-major f16 (B transposed)
    int M, int N, int K,
    float* __restrict__ outF32,       // used if outF16 == nullptr
    _Float16* __restrict__ outF16,    // used if != nullptr
    const float* __restrict__ bias,      // length N or nullptr
    const float* __restrict__ residual)  // M x N f32 or nullptr
{
  __shared__ __align__(16) _Float16 ldsA[128 * 32];
  __shared__ __align__(16) _Float16 ldsB[64 * 32];

  const int wave = threadIdx.x >> 5;
  const int lane = threadIdx.x & 31;
  const int bm   = blockIdx.x * 128;
  const int bn   = blockIdx.y * 64;
  const int wmr  = (wave & 3) * 32;   // wave row offset within block tile
  const int wnc  = (wave >> 2) * 32;  // wave col offset within block tile

  v8f acc[2][2] = {};

  for (int kk = 0; kk < K; kk += 32) {
    __syncthreads();   // previous tiles fully consumed
#if TDM_AVAILABLE
    if (wave == 0) {
      tdm_load_tile_2d(A  + (size_t)bm * K + kk,
                       (unsigned)(unsigned long long)(void*)ldsA, 128, 32, K);
      tdm_load_tile_2d(Bt + (size_t)bn * K + kk,
                       (unsigned)(unsigned long long)(void*)ldsB, 64, 32, K);
      __builtin_amdgcn_s_wait_tensorcnt(0);
    }
#else
    {
      const int t = threadIdx.x;
      // A tile: 128x32 halves = 512 16-byte chunks; 2 per thread
#pragma unroll
      for (int c = t; c < 512; c += 256) {
        int r = c >> 2, col = (c & 3) * 8;
        *(v8h*)(ldsA + r * 32 + col) =
            *(const v8h*)(A + (size_t)(bm + r) * K + kk + col);
      }
      // B tile: 64x32 halves = 256 16-byte chunks; 1 per thread
      {
        int r = t >> 2, col = (t & 3) * 8;
        *(v8h*)(ldsB + r * 32 + col) =
            *(const v8h*)(Bt + (size_t)(bn + r) * K + kk + col);
      }
    }
#endif
    __syncthreads();   // tiles visible to all waves

    v16h a0 = load_fragA(ldsA + (wmr + 0) * 32, 32);
    v16h a1 = load_fragA(ldsA + (wmr + 16) * 32, 32);
    v16h b0 = load_fragB(ldsB + (wnc + 0) * 32, 32);
    v16h b1 = load_fragB(ldsB + (wnc + 16) * 32, 32);
    acc[0][0] = wmma_f16(a0, b0, acc[0][0]);
    acc[0][1] = wmma_f16(a0, b1, acc[0][1]);
    acc[1][0] = wmma_f16(a1, b0, acc[1][0]);
    acc[1][1] = wmma_f16(a1, b1, acc[1][1]);
  }

  const int nlane = lane & 15;
  const int rbase = (lane >> 4) << 3;   // C-frag: lanes 0-15 -> M=j, 16-31 -> M=j+8
#pragma unroll
  for (int mi = 0; mi < 2; ++mi) {
#pragma unroll
    for (int ni = 0; ni < 2; ++ni) {
      const int c = bn + wnc + ni * 16 + nlane;
#pragma unroll
      for (int j = 0; j < 8; ++j) {
        const int r = bm + wmr + mi * 16 + rbase + j;
        float v = acc[mi][ni][j];
        if (bias)     v += bias[c];
        if (residual) v += residual[(size_t)r * N + c];
        if (outF16)   outF16[(size_t)r * N + c] = (_Float16)v;
        else          outF32[(size_t)r * N + c] = v;
      }
    }
  }
}

// ---------------------------------------------------------------------------
// Transpose V: [b*S+s, h*CH+ch] f16 -> Vt [(b*H+h)*CH+ch, s] f16
// ---------------------------------------------------------------------------
__global__ void v_transpose_kernel(const _Float16* __restrict__ V,
                                   _Float16* __restrict__ Vt) {
  int t = blockIdx.x * blockDim.x + threadIdx.x;
  if (t >= ROWS * CC) return;
  int c   = t & (CC - 1);
  int row = t >> 9;                 // b*S + s
  int b   = row >> 11;
  int s   = row & (SS - 1);
  int h   = c >> 6, ch = c & (CHD - 1);
  Vt[((size_t)(b * HH + h) * CHD + ch) * SS + s] = V[(size_t)row * CC + c];
}

// ---------------------------------------------------------------------------
// Flash attention: one wave per (b, h, 16-query tile). Online softmax.
// Q,K: [B*S, 512] f16 (head at col h*64).  Vt: [(b*H+h)*64+ch, S] f16.
// O:   [B*S, 512] f16.
// ---------------------------------------------------------------------------
__global__ __launch_bounds__(128) void attn_kernel(
    const _Float16* __restrict__ Q,
    const _Float16* __restrict__ Km,
    const _Float16* __restrict__ Vt,
    _Float16* __restrict__ O) {
  __shared__ __align__(16) _Float16 pbuf[4][16 * 32];

  const int wave  = threadIdx.x >> 5;
  const int lane  = threadIdx.x & 31;
  const int h     = blockIdx.y;
  const int b     = blockIdx.z;
  const int qt    = blockIdx.x * 4 + wave;      // 0..127
  const int qrow0 = b * SS + qt * 16;
  const int hoff  = h * CHD;

  // Q fragments: K-dim 64 -> two chunks of 32
  v16h q0 = load_fragA(Q + (size_t)qrow0 * CC + hoff, CC);
  v16h q1 = load_fragA(Q + (size_t)qrow0 * CC + hoff + 32, CC);

  float m[8], l[8];
#pragma unroll
  for (int j = 0; j < 8; ++j) { m[j] = -1e30f; l[j] = 0.f; }
  v8f o0 = {}, o1 = {}, o2 = {}, o3 = {};

  const _Float16* Vbase = Vt + (size_t)(b * HH + h) * CHD * SS;
  _Float16* P = pbuf[wave];
  const int nlane = lane & 15;
  const int rbase = (lane >> 4) << 3;

  for (int key0 = 0; key0 < SS; key0 += 32) {
    const _Float16* Kb = Km + (size_t)(b * SS + key0) * CC + hoff;
    // scores for 32 keys: two 16x16 tiles, each over K=64 (2 wmma)
    v8f s0 = {}, s1 = {};
    s0 = wmma_f16(q0, load_fragB(Kb, CC), s0);
    s0 = wmma_f16(q1, load_fragB(Kb + 32, CC), s0);
    s1 = wmma_f16(q0, load_fragB(Kb + (size_t)16 * CC, CC), s1);
    s1 = wmma_f16(q1, load_fragB(Kb + (size_t)16 * CC + 32, CC), s1);

    // online softmax per row (lane half holds rows rbase..rbase+7)
#pragma unroll
    for (int j = 0; j < 8; ++j) {
      float a0 = s0[j] * 0.125f;                 // 1/sqrt(64)
      float a1 = s1[j] * 0.125f;
      float mx = fmaxf(a0, a1);
#pragma unroll
      for (int d = 1; d < 16; d <<= 1) mx = fmaxf(mx, __shfl_xor(mx, d, 32));
      float newm  = fmaxf(m[j], mx);
      float alpha = __expf(m[j] - newm);
      float p0 = __expf(a0 - newm);
      float p1 = __expf(a1 - newm);
      float rs = p0 + p1;
#pragma unroll
      for (int d = 1; d < 16; d <<= 1) rs += __shfl_xor(rs, d, 32);
      l[j] = l[j] * alpha + rs;
      m[j] = newm;
      o0[j] *= alpha; o1[j] *= alpha; o2[j] *= alpha; o3[j] *= alpha;
      const int prow = rbase + j;
      P[prow * 32 + nlane]      = (_Float16)p0;
      P[prow * 32 + 16 + nlane] = (_Float16)p1;
    }

    // O += P(16x32) @ Vchunk(32x64): 4 wmma over CH chunks of 16
    v16h pf = load_fragA(P, 32);
    o0 = wmma_f16(pf, load_fragB(Vbase + (size_t)(0 * 16) * SS + key0, SS), o0);
    o1 = wmma_f16(pf, load_fragB(Vbase + (size_t)(1 * 16) * SS + key0, SS), o1);
    o2 = wmma_f16(pf, load_fragB(Vbase + (size_t)(2 * 16) * SS + key0, SS), o2);
    o3 = wmma_f16(pf, load_fragB(Vbase + (size_t)(3 * 16) * SS + key0, SS), o3);
  }

  // finalize: divide by l, write f16
#pragma unroll
  for (int j = 0; j < 8; ++j) {
    float inv = 1.0f / l[j];
    _Float16* op = O + (size_t)(qrow0 + rbase + j) * CC + hoff;
    op[0 * 16 + nlane] = (_Float16)(o0[j] * inv);
    op[1 * 16 + nlane] = (_Float16)(o1[j] * inv);
    op[2 * 16 + nlane] = (_Float16)(o2[j] * inv);
    op[3 * 16 + nlane] = (_Float16)(o3[j] * inv);
  }
}

// ---------------------------------------------------------------------------
// GELU-gate: g[row, 0:256]=a, g[row,256:512]=gate; ff = a * gelu_tanh(gate)
// ---------------------------------------------------------------------------
__global__ void act_kernel(const float* __restrict__ G, _Float16* __restrict__ FF) {
  int t = blockIdx.x * blockDim.x + threadIdx.x;
  if (t >= ROWS * ODD) return;
  int row = t >> 8, c = t & (ODD - 1);
  float a = G[(size_t)row * CC + c];
  float g = G[(size_t)row * CC + ODD + c];
  float th = tanhf(0.7978845608f * g * (1.0f + 0.044715f * g * g));
  FF[(size_t)row * ODD + c] = (_Float16)(a * 0.5f * g * (1.0f + th));
}

// ---------------------------------------------------------------------------
// Host-side orchestration
// ---------------------------------------------------------------------------
extern "C" void kernel_launch(void* const* d_in, const int* in_sizes, int n_in,
                              void* d_out, int out_size, void* d_ws, size_t ws_size,
                              hipStream_t stream) {
  (void)in_sizes; (void)n_in; (void)out_size; (void)ws_size;

  const float* x_in  = (const float*)d_in[0];
  const float* gamma = (const float*)d_in[1];   // (L+1, C)
  const float* beta  = (const float*)d_in[2];
  const float* Wq    = (const float*)d_in[3];   // (L, C, 512)
  const float* Wk    = (const float*)d_in[4];
  const float* Wv    = (const float*)d_in[5];
  const float* Wo    = (const float*)d_in[6];   // (L, 512, C)
  const float* bo    = (const float*)d_in[7];   // (L, C)
  const float* Wg    = (const float*)d_in[8];   // (C, 512)
  const float* bg    = (const float*)d_in[9];   // (512)
  const float* Wf    = (const float*)d_in[10];  // (256, C)
  const float* bf    = (const float*)d_in[11];  // (C)

  // workspace carve-up (256B aligned regions)
  char* ws = (char*)d_ws;
  auto alloc = [&](size_t bytes) -> char* {
    char* p = ws;
    ws += (bytes + 255) & ~(size_t)255;
    return p;
  };
  float*    xbuf = (float*)alloc((size_t)ROWS * CC * 4);
  float*    gbuf = (float*)alloc((size_t)ROWS * CC * 4);
  _Float16* h16  = (_Float16*)alloc((size_t)ROWS * CC * 2);
  _Float16* q16  = (_Float16*)alloc((size_t)ROWS * CC * 2);
  _Float16* k16  = (_Float16*)alloc((size_t)ROWS * CC * 2);
  _Float16* v16  = (_Float16*)alloc((size_t)ROWS * CC * 2);
  _Float16* vT16 = (_Float16*)alloc((size_t)ROWS * CC * 2);
  _Float16* o16  = (_Float16*)alloc((size_t)ROWS * CC * 2);
  _Float16* ff16 = (_Float16*)alloc((size_t)ROWS * ODD * 2);
  _Float16* WqT  = (_Float16*)alloc((size_t)LL * CC * CC * 2);
  _Float16* WkT  = (_Float16*)alloc((size_t)LL * CC * CC * 2);
  _Float16* WvT  = (_Float16*)alloc((size_t)LL * CC * CC * 2);
  _Float16* WoT  = (_Float16*)alloc((size_t)LL * CC * CC * 2);
  _Float16* WgT  = (_Float16*)alloc((size_t)CC * CC * 2);
  _Float16* WfT  = (_Float16*)alloc((size_t)CC * ODD * 2);

  // x working copy (x is updated per layer; inputs must not be mutated)
  hipMemcpyAsync(xbuf, x_in, (size_t)ROWS * CC * 4, hipMemcpyDeviceToDevice, stream);

  auto cvt = [&](const float* W, _Float16* Wt, int R, int N) {
    int n = R * N;
    cvt_transpose_kernel<<<(n + 255) / 256, 256, 0, stream>>>(W, Wt, R, N);
  };
  const size_t WSZ = (size_t)CC * CC;   // 262144
  for (int i = 0; i < LL; ++i) {
    cvt(Wq + i * WSZ, WqT + i * WSZ, CC, CC);
    cvt(Wk + i * WSZ, WkT + i * WSZ, CC, CC);
    cvt(Wv + i * WSZ, WvT + i * WSZ, CC, CC);
    cvt(Wo + i * WSZ, WoT + i * WSZ, CC, CC);
  }
  cvt(Wg, WgT, CC, CC);
  cvt(Wf, WfT, ODD, CC);

  const dim3 gGemm(ROWS / 128, CC / 64);        // 32 x 8 blocks
  const int  nElem = ROWS * CC;

  for (int i = 0; i < LL; ++i) {
    ln_kernel<<<ROWS / 8, 256, 0, stream>>>(xbuf, gamma + i * CC, beta + i * CC, h16);

    gemm_wmma_kernel<<<gGemm, 256, 0, stream>>>(h16, WqT + i * WSZ, ROWS, CC, CC,
                                                nullptr, q16, nullptr, nullptr);
    gemm_wmma_kernel<<<gGemm, 256, 0, stream>>>(h16, WkT + i * WSZ, ROWS, CC, CC,
                                                nullptr, k16, nullptr, nullptr);
    gemm_wmma_kernel<<<gGemm, 256, 0, stream>>>(h16, WvT + i * WSZ, ROWS, CC, CC,
                                                nullptr, v16, nullptr, nullptr);

    v_transpose_kernel<<<(nElem + 255) / 256, 256, 0, stream>>>(v16, vT16);

    attn_kernel<<<dim3(SS / 16 / 4, HH, BB), 128, 0, stream>>>(q16, k16, vT16, o16);

    // x = o @ Wo + bo + x   (residual in-place on xbuf)
    gemm_wmma_kernel<<<gGemm, 256, 0, stream>>>(o16, WoT + i * WSZ, ROWS, CC, CC,
                                                xbuf, nullptr, bo + i * CC, xbuf);
  }

  // MLP tail
  ln_kernel<<<ROWS / 8, 256, 0, stream>>>(xbuf, gamma + LL * CC, beta + LL * CC, h16);
  gemm_wmma_kernel<<<gGemm, 256, 0, stream>>>(h16, WgT, ROWS, CC, CC,
                                              gbuf, nullptr, bg, nullptr);
  act_kernel<<<(ROWS * ODD + 255) / 256, 256, 0, stream>>>(gbuf, ff16);
  gemm_wmma_kernel<<<gGemm, 256, 0, stream>>>(ff16, WfT, ROWS, CC, ODD,
                                              (float*)d_out, nullptr, bf, xbuf);
}
